// SingleRenderer_84524956385566
// MI455X (gfx1250) — matrix operand order repaired and checked by reading, not compiled, not run on hardware
//
#include <hip/hip_runtime.h>
#include <hip/hip_bf16.h>

typedef _Float16 h16 __attribute__((ext_vector_type(16)));
typedef _Float16 h8  __attribute__((ext_vector_type(8)));
typedef float    f8  __attribute__((ext_vector_type(8)));

#define NRAYS 8192
#define DSTR  128

// branch-free softplus using fast HW transcendentals:
// log(1+e^x) = max(x,0) + log(1 + e^{-|x|})
__device__ __forceinline__ float softplus_f(float x){
  return fmaxf(x, 0.f) + __logf(1.f + __expf(-fabsf(x)));
}
__device__ __forceinline__ float sigmoid_f(float x){
  return 1.f / (1.f + __expf(-x));
}

// A fragment (16x32 f16) from LDS row-major activations act[16][64].
// ISA layout: lane%16 = M row; lane/16 selects 8-wide K chunks {0..7,16..23} vs {8..15,24..31}.
__device__ __forceinline__ h16 load_afrag(const _Float16* act, int kt, int lane){
  int m  = lane & 15;
  int hi = (lane >> 4) * 8;
  const _Float16* row = act + m*64 + kt*32;
  const h8* p0 = (const h8*)(row + hi);
  const h8* p1 = (const h8*)(row + 16 + hi);
  h16 a;
#pragma unroll
  for(int e=0;e<8;e++){ a[e] = (*p0)[e]; a[8+e] = (*p1)[e]; }
  return a;
}

// B fragment (32x16 f16) from pre-packed weights WT[N][64] row-major
// (lane%16 = column N, lane/16 selects 8-wide K chunks {0..7,16..23} vs {8..15,24..31}).
__device__ __forceinline__ h16 load_bfrag(const _Float16* WT, int nt, int kt, int lane){
  int n = nt*16 + (lane & 15);
  int k = kt*32 + (lane >> 4)*8;
  const h8* p0 = (const h8*)(WT + n*64 + k);
  const h8* p1 = (const h8*)(WT + n*64 + k + 16);
  h16 b;
#pragma unroll
  for(int e=0;e<8;e++){ b[e] = (*p0)[e]; b[8+e] = (*p1)[e]; }
  return b;
}

// ---------------- weight pack: f16, B-fragment-friendly ----------------
__global__ void prep_kernel(const float* W2, const float* RW2,
                            _Float16* W2T, _Float16* W2R, _Float16* RW2T){
  int idx = blockIdx.x*blockDim.x + threadIdx.x;
  if(idx >= 64*64) return;
  int n = idx >> 6, k = idx & 63;
  W2T[idx]  = (_Float16)W2[k*64 + n];   // forward:  B[k][n] = W2[k][n]  -> pack[n][k]
  W2R[idx]  = (_Float16)W2[idx];        // backward: B[j][i] = W2[i][j]  -> pack[i][j]
  RW2T[idx] = (_Float16)RW2[k*64 + n];
}

// ---------------- ray setup + coarse samples ----------------
__global__ void setup_kernel(const float* rays_o, const float* rays_d,
                             float* dirn, float* d_all){
  int ray = blockIdx.x*blockDim.x + threadIdx.x;
  if(ray >= NRAYS) return;
  float ox=rays_o[ray*3+0], oy=rays_o[ray*3+1], oz=rays_o[ray*3+2];
  float dx=rays_d[ray*3+0], dy=rays_d[ray*3+1], dz=rays_d[ray*3+2];
  float inv = rsqrtf(dx*dx+dy*dy+dz*dz);
  dx*=inv; dy*=inv; dz*=inv;
  dirn[ray*3+0]=dx; dirn[ray*3+1]=dy; dirn[ray*3+2]=dz;
  float mid = -(ox*dx+oy*dy+oz*dz);
  float nearv = fmaxf(mid-1.f, 0.01f);
  float farv  = mid+1.f;
#pragma unroll 4
  for(int t=0;t<64;t++){
    float u = (float)t * (1.f/63.f);
    d_all[ray*DSTR + t] = nearv*(1.f-u) + farv*u;
  }
}

// ---------------- SDF forward (coarse / fine): 32 points per wave ----------------
__global__ __launch_bounds__(256)
void sdf_eval_kernel(const float* rays_o, const float* dirn,
                     const float* dbuf, int dstride, int S,
                     float* sdfbuf, int sstride,
                     const float* W1, const float* b1,
                     const _Float16* W2T, const float* b2,
                     const float* W3, const float* b3)
{
  __shared__ __align__(32) _Float16 h1s[8][2048];   // 8 waves x (32 pts x 64)
  int lane = threadIdx.x & 31;
  int wave = threadIdx.x >> 5;
  int base = (blockIdx.x*8 + wave) * 32;
  _Float16* myh1 = h1s[wave];
  int j0 = (lane>>4)*32;
  int n0 = lane & 15;
#pragma unroll
  for(int half=0; half<2; half++){
    int m = n0 + half*16;
    int pidx = base + m;
    int ray = pidx / S;
    int smp = pidx - ray*S;
    float ox=rays_o[ray*3], oy=rays_o[ray*3+1], oz=rays_o[ray*3+2];
    float dx=dirn[ray*3],  dy=dirn[ray*3+1],  dz=dirn[ray*3+2];
    float t = dbuf[ray*dstride + smp];
    float px=ox+t*dx, py=oy+t*dy, pz=oz+t*dz;
#pragma unroll 8
    for(int jj=0;jj<32;jj++){
      int j=j0+jj;
      float v = px*W1[j] + py*W1[64+j] + pz*W1[128+j] + b1[j];
      myh1[m*64+j] = (_Float16)softplus_f(v);
    }
  }
  __syncthreads();
  h16 a00 = load_afrag(myh1,        0, lane);
  h16 a01 = load_afrag(myh1,        1, lane);
  h16 a10 = load_afrag(myh1+16*64,  0, lane);
  h16 a11 = load_afrag(myh1+16*64,  1, lane);
  f8 part0, part1;
#pragma unroll
  for(int r=0;r<8;r++){ part0[r]=0.f; part1[r]=0.f; }
#pragma unroll
  for(int nt=0;nt<4;nt++){
    float bv = b2[nt*16 + n0];
    f8 c0, c1;
#pragma unroll
    for(int r=0;r<8;r++){ c0[r]=bv; c1[r]=bv; }
    h16 b0 = load_bfrag(W2T, nt, 0, lane);
    h16 b1 = load_bfrag(W2T, nt, 1, lane);
    c0 = __builtin_amdgcn_wmma_f32_16x16x32_f16(false, a00, false, b0, (short)0, c0, false, false);
    c0 = __builtin_amdgcn_wmma_f32_16x16x32_f16(false, a01, false, b1, (short)0, c0, false, false);
    c1 = __builtin_amdgcn_wmma_f32_16x16x32_f16(false, a10, false, b0, (short)0, c1, false, false);
    c1 = __builtin_amdgcn_wmma_f32_16x16x32_f16(false, a11, false, b1, (short)0, c1, false, false);
    float w3 = W3[nt*16 + n0];
#pragma unroll
    for(int r=0;r<8;r++){
      part0[r] += softplus_f(c0[r]) * w3;
      part1[r] += softplus_f(c1[r]) * w3;
    }
  }
#pragma unroll
  for(int mask=1; mask<16; mask<<=1){
#pragma unroll
    for(int r=0;r<8;r++){
      part0[r] += __shfl_xor(part0[r], mask, 32);
      part1[r] += __shfl_xor(part1[r], mask, 32);
    }
  }
  if(n0==0){
    int mb = (lane>>4)*8;
    float bb = b3[0];
#pragma unroll
    for(int r=0;r<8;r++){
      int pi = base + mb + r;
      int rr = pi / S, ss = pi - rr*S;
      sdfbuf[rr*sstride + ss] = part0[r] + bb;
    }
#pragma unroll
    for(int r=0;r<8;r++){
      int pi = base + 16 + mb + r;
      int rr = pi / S, ss = pi - rr*S;
      sdfbuf[rr*sstride + ss] = part1[r] + bb;
    }
  }
}

// ---------------- importance sampling (1 thread / ray) ----------------
__global__ void upsample_kernel(const float* d_all, const float* sdf_all,
                                float* dfine, float* cdfbuf, int n, float s_i)
{
  int ray = blockIdx.x*blockDim.x + threadIdx.x;
  if(ray >= NRAYS) return;
  const float* d  = d_all   + ray*DSTR;
  const float* sd = sdf_all + ray*DSTR;
  float* cdf = cdfbuf + ray*DSTR;
  float T = 1.f, total = 0.f, prev_dot = 0.f;
  for(int k=0;k<n-1;k++){
    float ps=sd[k], ns=sd[k+1], pz=d[k], nz=d[k+1];
    float mid_sdf = 0.5f*(ps+ns);
    float dv = (ns-ps)/(nz-pz+1e-5f);
    float dcl = fminf(prev_dot, dv);
    dcl = fminf(fmaxf(dcl,-10.f), 0.f);
    prev_dot = dv;
    float dist = nz-pz;
    float pc = sigmoid_f((mid_sdf - dcl*dist*0.5f)*s_i);
    float nc = sigmoid_f((mid_sdf + dcl*dist*0.5f)*s_i);
    float alpha = (pc-nc+1e-5f)/(pc+1e-5f);
    float wv = alpha*T;
    T *= (1.f-alpha+1e-10f);
    float wp = wv + 1e-5f;
    cdf[k] = wp; total += wp;
  }
  float inv_total = 1.f/total;
  float run = 0.f, wprev = cdf[0];
  cdf[0] = 0.f;
  for(int k=1;k<n;k++){
    run += wprev*inv_total;
    float nxt = (k<=n-2)? cdf[k] : 0.f;
    cdf[k] = run;
    wprev = nxt;
  }
  int ind = 1;
  for(int t=0;t<16;t++){
    float u = (float)t * (1.f/15.f);
    while(ind < n && cdf[ind] <= u) ind++;
    int below = ind-1; if(below > n-1) below = n-1;
    int above = (ind < n-1)? ind : n-1;
    float c0=cdf[below], c1=cdf[above];
    float b0v=d[below],  b1v=d[above];
    float den = c1-c0; den = (den < 1e-5f)? 1.f : den;
    float tt = (u-c0)/den;
    dfine[ray*16+t] = b0v + tt*(b1v-b0v);
  }
}

// in-place backward merge of sorted (d,sdf) with 16 fine samples (stable: fine after on ties)
__global__ void merge_kernel(float* d_all, float* sdf_all,
                             const float* dfine, const float* sdffine, int n)
{
  int ray = blockIdx.x*blockDim.x + threadIdx.x;
  if(ray >= NRAYS) return;
  float* d = d_all + ray*DSTR;
  float* s = sdf_all + ray*DSTR;
  const float* df = dfine + ray*16;
  const float* sf = sdffine + ray*16;
  int i = n-1, j = 15;
  for(int w = n+15; w >= 0; w--){
    bool takeMain = (j < 0) || (i >= 0 && d[i] > df[j]);
    if(takeMain){ d[w]=d[i]; s[w]=s[i]; i--; }
    else        { d[w]=df[j]; s[w]=sf[j]; j--; }
  }
}

// ---------------- final SDF forward + VJP (normals): 32 points/wave, 4 waves ----------------
__global__ __launch_bounds__(128)
void sdf_final_kernel(const float* rays_o, const float* dirn, const float* d_all,
                      float* sdfbuf, float* nabla,
                      const float* W1, const float* b1,
                      const _Float16* W2T, const float* b2,
                      const float* W3, const float* b3,
                      const _Float16* W2R)
{
  __shared__ __align__(32) _Float16 h1s[4][2048];
  __shared__ __align__(32) _Float16 h2s[4][2048];
  __shared__ __align__(32) _Float16 g1s[4][2048];
  int lane = threadIdx.x & 31;
  int wave = threadIdx.x >> 5;
  int base = (blockIdx.x*4 + wave) * 32;
  _Float16* myh1 = h1s[wave];
  _Float16* myh2 = h2s[wave];
  _Float16* myg1 = g1s[wave];
  int j0 = (lane>>4)*32;
  int n0 = lane & 15;
  int mrow = (lane>>4)*8;
#pragma unroll
  for(int half=0; half<2; half++){
    int m = n0 + half*16;
    int pidx = base + m;
    int ray = pidx >> 7;
    int smp = pidx & 127;
    float ox=rays_o[ray*3], oy=rays_o[ray*3+1], oz=rays_o[ray*3+2];
    float dx=dirn[ray*3],  dy=dirn[ray*3+1],  dz=dirn[ray*3+2];
    float t = d_all[ray*DSTR + smp];
    float px=ox+t*dx, py=oy+t*dy, pz=oz+t*dz;
#pragma unroll 8
    for(int jj=0;jj<32;jj++){
      int j=j0+jj;
      float v = px*W1[j] + py*W1[64+j] + pz*W1[128+j] + b1[j];
      myh1[m*64+j] = (_Float16)softplus_f(v);
    }
  }
  __syncthreads();
  // forward layer 2 + fused layer 3 + h2 stash
  {
    h16 a00 = load_afrag(myh1,       0, lane);
    h16 a01 = load_afrag(myh1,       1, lane);
    h16 a10 = load_afrag(myh1+16*64, 0, lane);
    h16 a11 = load_afrag(myh1+16*64, 1, lane);
    f8 part0, part1;
#pragma unroll
    for(int r=0;r<8;r++){ part0[r]=0.f; part1[r]=0.f; }
#pragma unroll
    for(int nt=0;nt<4;nt++){
      float bv = b2[nt*16 + n0];
      f8 c0, c1;
#pragma unroll
      for(int r=0;r<8;r++){ c0[r]=bv; c1[r]=bv; }
      h16 b0 = load_bfrag(W2T, nt, 0, lane);
      h16 b1 = load_bfrag(W2T, nt, 1, lane);
      c0 = __builtin_amdgcn_wmma_f32_16x16x32_f16(false, a00, false, b0, (short)0, c0, false, false);
      c0 = __builtin_amdgcn_wmma_f32_16x16x32_f16(false, a01, false, b1, (short)0, c0, false, false);
      c1 = __builtin_amdgcn_wmma_f32_16x16x32_f16(false, a10, false, b0, (short)0, c1, false, false);
      c1 = __builtin_amdgcn_wmma_f32_16x16x32_f16(false, a11, false, b1, (short)0, c1, false, false);
      float w3 = W3[nt*16 + n0];
#pragma unroll
      for(int r=0;r<8;r++){
        float h0 = softplus_f(c0[r]);
        float h1v = softplus_f(c1[r]);
        myh2[(mrow+r)*64      + nt*16 + n0] = (_Float16)h0;
        myh2[(16+mrow+r)*64   + nt*16 + n0] = (_Float16)h1v;
        part0[r] += h0*w3;
        part1[r] += h1v*w3;
      }
    }
#pragma unroll
    for(int mask=1; mask<16; mask<<=1){
#pragma unroll
      for(int r=0;r<8;r++){
        part0[r] += __shfl_xor(part0[r], mask, 32);
        part1[r] += __shfl_xor(part1[r], mask, 32);
      }
    }
    if(n0==0){
      float bb = b3[0];
#pragma unroll
      for(int r=0;r<8;r++){
        int pi = base + mrow + r;
        sdfbuf[(pi>>7)*DSTR + (pi&127)] = part0[r] + bb;
      }
#pragma unroll
      for(int r=0;r<8;r++){
        int pi = base + 16 + mrow + r;
        sdfbuf[(pi>>7)*DSTR + (pi&127)] = part1[r] + bb;
      }
    }
  }
  __syncthreads();
  // backward: A2 = W3 * sigma'(z2),  sigma'(z) = 1 - exp(-softplus(z))
  h16 a20, a21, a30, a31;
  {
    int hi = (lane>>4)*8;
#pragma unroll
    for(int mt=0; mt<2; mt++){
      const _Float16* row = myh2 + (mt*16 + n0)*64;
#pragma unroll
      for(int kt=0;kt<2;kt++){
        h16 a;
#pragma unroll
        for(int e=0;e<16;e++){
          int k = kt*32 + (e&7) + ((e>>3)*16) + hi;
          float hv = (float)row[k];
          a[e] = (_Float16)(W3[k]*(1.f - __expf(-hv)));
        }
        if(mt==0){ if(kt==0) a20=a; else a21=a; }
        else     { if(kt==0) a30=a; else a31=a; }
      }
    }
  }
#pragma unroll
  for(int nt=0;nt<4;nt++){
    f8 c0, c1;
#pragma unroll
    for(int r=0;r<8;r++){ c0[r]=0.f; c1[r]=0.f; }
    h16 b0 = load_bfrag(W2R, nt, 0, lane);
    h16 b1 = load_bfrag(W2R, nt, 1, lane);
    c0 = __builtin_amdgcn_wmma_f32_16x16x32_f16(false, a20, false, b0, (short)0, c0, false, false);
    c0 = __builtin_amdgcn_wmma_f32_16x16x32_f16(false, a21, false, b1, (short)0, c0, false, false);
    c1 = __builtin_amdgcn_wmma_f32_16x16x32_f16(false, a30, false, b0, (short)0, c1, false, false);
    c1 = __builtin_amdgcn_wmma_f32_16x16x32_f16(false, a31, false, b1, (short)0, c1, false, false);
    int i = nt*16 + n0;
#pragma unroll
    for(int r=0;r<8;r++){
      float hA = (float)myh1[(mrow+r)*64 + i];
      float hB = (float)myh1[(16+mrow+r)*64 + i];
      myg1[(mrow+r)*64 + i]    = (_Float16)(c0[r] * (1.f - __expf(-hA)));
      myg1[(16+mrow+r)*64 + i] = (_Float16)(c1[r] * (1.f - __expf(-hB)));
    }
  }
  __syncthreads();
  // nabla[c] = sum_j W1[c][j] * g1s[j]   — one lane per point (32 points / 32 lanes)
  {
    float a0=0.f, a1=0.f, a2=0.f;
    const _Float16* grow = myg1 + lane*64;
#pragma unroll 8
    for(int j=0;j<64;j++){
      float g = (float)grow[j];
      a0 += W1[j]*g; a1 += W1[64+j]*g; a2 += W1[128+j]*g;
    }
    int pidx = base + lane;
    float* np = nabla + ((size_t)((pidx>>7)*DSTR + (pidx&127)))*3;
    np[0]=a0; np[1]=a1; np[2]=a2;
  }
}

// ---------------- radiance MLP at midpoints: 32 points per wave ----------------
__global__ __launch_bounds__(256)
void rad_kernel(const float* rays_o, const float* dirn, const float* d_all,
                float* rad,
                const float* RW1, const float* Rb1,
                const _Float16* RW2T, const float* Rb2,
                const float* RW3, const float* Rb3)
{
  __shared__ __align__(32) _Float16 h1s[8][2048];
  int lane = threadIdx.x & 31;
  int wave = threadIdx.x >> 5;
  int base = (blockIdx.x*8 + wave)*32;
  _Float16* myh1 = h1s[wave];
  int j0 = (lane>>4)*32;
  int n0 = lane & 15;
#pragma unroll
  for(int half=0; half<2; half++){
    int m = n0 + half*16;
    int pidx = base + m;
    int ray = pidx / 127;
    int smp = pidx - ray*127;
    float ox=rays_o[ray*3], oy=rays_o[ray*3+1], oz=rays_o[ray*3+2];
    float dx=dirn[ray*3],  dy=dirn[ray*3+1],  dz=dirn[ray*3+2];
    float dm = 0.5f*(d_all[ray*DSTR+smp] + d_all[ray*DSTR+smp+1]);
    float px=ox+dm*dx, py=oy+dm*dy, pz=oz+dm*dz;
#pragma unroll 8
    for(int jj=0;jj<32;jj++){
      int j=j0+jj;
      float v = px*RW1[j] + py*RW1[64+j] + pz*RW1[128+j]
              + dx*RW1[192+j] + dy*RW1[256+j] + dz*RW1[320+j] + Rb1[j];
      myh1[m*64+j] = (_Float16)fmaxf(v, 0.f);
    }
  }
  __syncthreads();
  h16 a00 = load_afrag(myh1,       0, lane);
  h16 a01 = load_afrag(myh1,       1, lane);
  h16 a10 = load_afrag(myh1+16*64, 0, lane);
  h16 a11 = load_afrag(myh1+16*64, 1, lane);
  f8 p0a,p1a,p2a,p0b,p1b,p2b;
#pragma unroll
  for(int r=0;r<8;r++){ p0a[r]=0.f; p1a[r]=0.f; p2a[r]=0.f; p0b[r]=0.f; p1b[r]=0.f; p2b[r]=0.f; }
#pragma unroll
  for(int nt=0;nt<4;nt++){
    float bv = Rb2[nt*16 + n0];
    f8 c0, c1;
#pragma unroll
    for(int r=0;r<8;r++){ c0[r]=bv; c1[r]=bv; }
    h16 b0 = load_bfrag(RW2T, nt, 0, lane);
    h16 b1 = load_bfrag(RW2T, nt, 1, lane);
    c0 = __builtin_amdgcn_wmma_f32_16x16x32_f16(false, a00, false, b0, (short)0, c0, false, false);
    c0 = __builtin_amdgcn_wmma_f32_16x16x32_f16(false, a01, false, b1, (short)0, c0, false, false);
    c1 = __builtin_amdgcn_wmma_f32_16x16x32_f16(false, a10, false, b0, (short)0, c1, false, false);
    c1 = __builtin_amdgcn_wmma_f32_16x16x32_f16(false, a11, false, b1, (short)0, c1, false, false);
    int i = nt*16 + n0;
    float w0=RW3[i*3+0], w1=RW3[i*3+1], w2=RW3[i*3+2];
#pragma unroll
    for(int r=0;r<8;r++){
      float h0 = fmaxf(c0[r], 0.f);
      float h1v = fmaxf(c1[r], 0.f);
      p0a[r]+=h0*w0; p1a[r]+=h0*w1; p2a[r]+=h0*w2;
      p0b[r]+=h1v*w0; p1b[r]+=h1v*w1; p2b[r]+=h1v*w2;
    }
  }
#pragma unroll
  for(int mask=1; mask<16; mask<<=1){
#pragma unroll
    for(int r=0;r<8;r++){
      p0a[r]+=__shfl_xor(p0a[r], mask, 32);
      p1a[r]+=__shfl_xor(p1a[r], mask, 32);
      p2a[r]+=__shfl_xor(p2a[r], mask, 32);
      p0b[r]+=__shfl_xor(p0b[r], mask, 32);
      p1b[r]+=__shfl_xor(p1b[r], mask, 32);
      p2b[r]+=__shfl_xor(p2b[r], mask, 32);
    }
  }
  if(n0==0){
    int mb=(lane>>4)*8;
    float bb0=Rb3[0], bb1=Rb3[1], bb2=Rb3[2];
#pragma unroll
    for(int r=0;r<8;r++){
      int pi = base + mb + r;
      int rr = pi / 127, ss = pi - rr*127;
      float* o = rad + ((size_t)(rr*DSTR + ss))*3;
      o[0]=sigmoid_f(p0a[r]+bb0);
      o[1]=sigmoid_f(p1a[r]+bb1);
      o[2]=sigmoid_f(p2a[r]+bb2);
    }
#pragma unroll
    for(int r=0;r<8;r++){
      int pi = base + 16 + mb + r;
      int rr = pi / 127, ss = pi - rr*127;
      float* o = rad + ((size_t)(rr*DSTR + ss))*3;
      o[0]=sigmoid_f(p0b[r]+bb0);
      o[1]=sigmoid_f(p1b[r]+bb1);
      o[2]=sigmoid_f(p2b[r]+bb2);
    }
  }
}

// ---------------- compositing ----------------
__global__ void composite_kernel(const float* d_all, const float* sdf_all,
                                 const float* nabla, const float* rad,
                                 const float* ln_s, float* out)
{
  int ray = blockIdx.x*blockDim.x + threadIdx.x;
  if(ray >= NRAYS) return;
  float s = __expf(10.f * ln_s[0]);
  const float* d  = d_all   + ray*DSTR;
  const float* sd = sdf_all + ray*DSTR;
  float T=1.f, acc=0.f, dsum=0.f;
  float r0=0.f,r1=0.f,r2=0.f, n0=0.f,n1=0.f,n2=0.f;
  float cprev = sigmoid_f(sd[0]*s);
  for(int k=0;k<127;k++){
    float cnext = sigmoid_f(sd[k+1]*s);
    float alpha = fmaxf((cprev-cnext)/(cprev+1e-10f), 0.f);
    float wv = alpha*T;
    T *= (1.f-alpha+1e-10f);
    cprev = cnext;
    float dmid = 0.5f*(d[k]+d[k+1]);
    acc += wv; dsum += wv*dmid;
    const float* rp = rad + ((size_t)(ray*DSTR + k))*3;
    r0 += wv*rp[0]; r1 += wv*rp[1]; r2 += wv*rp[2];
    const float* np = nabla + ((size_t)(ray*DSTR + k))*3;
    float nx=np[0], ny=np[1], nz=np[2];
    float nrm = fmaxf(sqrtf(nx*nx+ny*ny+nz*nz), 1e-12f);
    float iw = wv/nrm;
    n0 += iw*nx; n1 += iw*ny; n2 += iw*nz;
  }
  float depth = dsum/(acc+1e-10f);
  float nl = fmaxf(sqrtf(n0*n0+n1*n1+n2*n2), 1e-12f);
  float* o = out + ray*8;
  o[0]=r0; o[1]=r1; o[2]=r2;
  o[3]=depth; o[4]=acc;
  o[5]=n0/nl; o[6]=n1/nl; o[7]=n2/nl;
}

extern "C" void kernel_launch(void* const* d_in, const int* in_sizes, int n_in,
                              void* d_out, int out_size, void* d_ws, size_t ws_size,
                              hipStream_t stream){
  (void)in_sizes; (void)n_in; (void)out_size; (void)ws_size;
  const float* rays_o=(const float*)d_in[0];
  const float* rays_d=(const float*)d_in[1];
  const float* W1 =(const float*)d_in[2];
  const float* b1 =(const float*)d_in[3];
  const float* W2 =(const float*)d_in[4];
  const float* b2 =(const float*)d_in[5];
  const float* W3 =(const float*)d_in[6];
  const float* b3 =(const float*)d_in[7];
  const float* RW1=(const float*)d_in[8];
  const float* Rb1=(const float*)d_in[9];
  const float* RW2=(const float*)d_in[10];
  const float* Rb2=(const float*)d_in[11];
  const float* RW3=(const float*)d_in[12];
  const float* Rb3=(const float*)d_in[13];
  const float* ln_s=(const float*)d_in[14];
  float* out=(float*)d_out;

  char* wsp = (char*)d_ws;
  size_t off = 0;
  auto take=[&](size_t bytes)->char*{
    char* p = wsp + off;
    off = (off + bytes + 255) & ~(size_t)255;
    return p;
  };
  float* d_all   = (float*)take((size_t)NRAYS*DSTR*4);
  float* sdf_all = (float*)take((size_t)NRAYS*DSTR*4);
  float* d_fine  = (float*)take((size_t)NRAYS*16*4);
  float* sdf_fine= (float*)take((size_t)NRAYS*16*4);
  float* cdfbuf  = (float*)take((size_t)NRAYS*DSTR*4);
  float* nabla   = (float*)take((size_t)NRAYS*DSTR*3*4);
  float* rad     = (float*)take((size_t)NRAYS*DSTR*3*4);
  float* dirn    = (float*)take((size_t)NRAYS*3*4);
  _Float16* W2T  = (_Float16*)take(64*64*2);
  _Float16* W2R  = (_Float16*)take(64*64*2);
  _Float16* RW2T = (_Float16*)take(64*64*2);

  prep_kernel<<<16,256,0,stream>>>(W2, RW2, W2T, W2R, RW2T);
  setup_kernel<<<32,256,0,stream>>>(rays_o, rays_d, dirn, d_all);
  sdf_eval_kernel<<<NRAYS*64/256,256,0,stream>>>(rays_o,dirn,d_all,DSTR,64,sdf_all,DSTR,
                                                 W1,b1,W2T,b2,W3,b3);
  int n = 64;
  for(int i=0;i<4;i++){
    upsample_kernel<<<32,256,0,stream>>>(d_all,sdf_all,d_fine,cdfbuf,n,64.f*(float)(1<<i));
    sdf_eval_kernel<<<NRAYS*16/256,256,0,stream>>>(rays_o,dirn,d_fine,16,16,sdf_fine,16,
                                                   W1,b1,W2T,b2,W3,b3);
    merge_kernel<<<32,256,0,stream>>>(d_all,sdf_all,d_fine,sdf_fine,n);
    n += 16;
  }
  sdf_final_kernel<<<NRAYS,128,0,stream>>>(rays_o,dirn,d_all,sdf_all,nabla,
                                           W1,b1,W2T,b2,W3,b3,W2R);
  rad_kernel<<<NRAYS*127/256,256,0,stream>>>(rays_o,dirn,d_all,rad,
                                             RW1,Rb1,RW2T,Rb2,RW3,Rb3);
  composite_kernel<<<32,256,0,stream>>>(d_all,sdf_all,nabla,rad,ln_s,out);
}
